// ClassifierMoE_50998441673011
// MI455X (gfx1250) — matrix operand order repaired and checked by reading, not compiled, not run on hardware
//
#include <hip/hip_runtime.h>

#define D_IN  2048
#define H1    512
#define H2    128
#define N_EXP 8
#define N_CLS 10
#define BATCH 16384

typedef __attribute__((ext_vector_type(16))) __bf16 v16bf;
typedef __attribute__((ext_vector_type(8)))  float  v8f;

__device__ __forceinline__ unsigned short f2bf(float x) {
  union { float f; unsigned int u; } v; v.f = x;
  unsigned int u = v.u;
  u += 0x7FFFu + ((u >> 16) & 1u);   // round-to-nearest-even
  return (unsigned short)(u >> 16);
}

union Frag {
  v16bf v;
  uint4 q[2];
};

// ---------------------------------------------------------------------------
// Pack features: concat(fs, fp) f32 -> bf16 row-major X[B, D_IN]
// ---------------------------------------------------------------------------
__global__ void pack_features_k(const float* __restrict__ fs,
                                const float* __restrict__ fp,
                                unsigned short* __restrict__ xb) {
  size_t idx = (size_t)blockIdx.x * blockDim.x + threadIdx.x;
  if (idx >= (size_t)BATCH * D_IN) return;
  int d = (int)(idx & (D_IN - 1));
  size_t b = idx >> 11;
  float v = (d < D_IN / 2) ? fs[b * (D_IN / 2) + d]
                           : fp[b * (D_IN / 2) + (d - D_IN / 2)];
  xb[idx] = f2bf(v);
}

// ---------------------------------------------------------------------------
// Pack a row-major [K x N] f32 weight matrix into WMMA B-fragment order (bf16).
// Fragment (kt, nt): 32 lanes x 16 bf16, contiguous per lane.
//   lane holds column n = nt*16 + (lane % 16)
//   element i holds   k = kt*32 + (i/8)*16 + (lane/16)*8 + (i%8)
// blockIdx.y batches stacked expert weights.
// ---------------------------------------------------------------------------
__global__ void pack_weight_k(const float* __restrict__ W,
                              unsigned short* __restrict__ out,
                              int K, int N, int Npad) {
  int total = K * Npad;
  int idx = blockIdx.x * blockDim.x + threadIdx.x;
  if (idx >= total) return;
  const float* Wz = W + (size_t)blockIdx.y * K * N;
  unsigned short* oz = out + (size_t)blockIdx.y * total;
  int i    = idx & 15;
  int lane = (idx >> 4) & 31;
  int frag = idx >> 9;
  int nTiles = Npad >> 4;
  int kt = frag / nTiles;
  int nt = frag - kt * nTiles;
  int n = nt * 16 + (lane & 15);
  int k = kt * 32 + ((i >> 3) << 4) + ((lane >> 4) << 3) + (i & 7);
  float v = (n < N) ? Wz[(size_t)k * N + n] : 0.0f;
  oz[idx] = f2bf(v);
}

// ---------------------------------------------------------------------------
// C = act(A[B,K] @ W + bias)  using v_wmma_f32_16x16x32_bf16.
// 8 waves / block along M share one N-panel of B-fragments, staged in LDS
// (double-buffered, one barrier per K-tile).  blockIdx.y = N-group,
// blockIdx.z = expert batch.
// ---------------------------------------------------------------------------
template <int MTILES, int NTILES, bool RELU, bool OUT_F32>
__global__ __launch_bounds__(256) void gemm_bf16_wmma_k(
    const unsigned short* __restrict__ A, size_t aStrideZ, int K,
    const unsigned short* __restrict__ Wp, size_t wStrideZ,
    const float* __restrict__ bias, int biasStrideZ,
    void* __restrict__ Cout, size_t cStrideZ, int ldc,
    int Npad, int Nact) {
  const int lane = threadIdx.x & 31;
  const int wave = threadIdx.x >> 5;
  const int mtBase = (blockIdx.x * 8 + wave) * MTILES;
  const int ng = blockIdx.y;
  const int z  = blockIdx.z;

  const unsigned short* Az = A  + (size_t)z * aStrideZ;
  const unsigned short* Wz = Wp + (size_t)z * wStrideZ;
  const float*          bz = bias + (size_t)z * biasStrideZ;

  const int sel    = lane >> 4;          // which half-wave (A/B K sub-block)
  const int nTiles = Npad >> 4;
  const int kTiles = K >> 5;

  // Shared B panel: NTILES fragments x (32 lanes x 32B) per stage, 2 stages.
  __shared__ alignas(16) unsigned short Bsh[2][NTILES * 512];

  // A fragment source rows (ISA 16-bit A 16x32 layout):
  // lane holds row M = lane%16; chunk0 = K[sel*8 .. +8), chunk1 = K[16+sel*8 .. +8)
  const unsigned short* arow[MTILES];
#pragma unroll
  for (int mi = 0; mi < MTILES; ++mi)
    arow[mi] = Az + (size_t)((mtBase + mi) * 16 + (lane & 15)) * K + sel * 8;

  // Cooperative B-panel load: NTILES KB contiguous per K-tile, 16B per thread.
  const int tid = threadIdx.x;
  auto coopLoadB = [&](int kt, int stage) {
    if (tid * 16 < NTILES * 1024) {
      const uint4* src =
          (const uint4*)(Wz + ((size_t)kt * nTiles + ng * NTILES) * 512) + tid;
      ((uint4*)&Bsh[stage][0])[tid] = *src;
    }
  };

  v8f acc[MTILES][NTILES];
#pragma unroll
  for (int mi = 0; mi < MTILES; ++mi)
#pragma unroll
    for (int t = 0; t < NTILES; ++t) acc[mi][t] = {};

  coopLoadB(0, 0);
  __syncthreads();

  for (int kt = 0; kt < kTiles; ++kt) {
    const int cur = kt & 1;
    if (kt + 1 < kTiles) coopLoadB(kt + 1, cur ^ 1);   // prefetch next panel
    if (kt + 8 < kTiles) {                             // stream-prefetch A rows
#pragma unroll
      for (int mi = 0; mi < MTILES; ++mi)
        __builtin_prefetch((const void*)(arow[mi] + (kt + 8) * 32), 0, 1);
    }

    Frag a[MTILES];
#pragma unroll
    for (int mi = 0; mi < MTILES; ++mi) {
      a[mi].q[0] = *(const uint4*)(arow[mi] + kt * 32);
      a[mi].q[1] = *(const uint4*)(arow[mi] + kt * 32 + 16);
    }
#pragma unroll
    for (int t = 0; t < NTILES; ++t) {
      const unsigned short* bbase = &Bsh[cur][0] + t * 512 + lane * 16;
      Frag b;
      b.q[0] = *(const uint4*)(bbase);
      b.q[1] = *(const uint4*)(bbase + 8);
#pragma unroll
      for (int mi = 0; mi < MTILES; ++mi) {
        acc[mi][t] = __builtin_amdgcn_wmma_f32_16x16x32_bf16(
            false, a[mi].v, false, b.v, (short)0, acc[mi][t], false, false);
      }
    }
    __syncthreads();   // protects the stage written this iter (read last iter)
  }

  // C/D layout: lane -> N = lane%16; reg r -> M = (lane/16)*8 + r
#pragma unroll
  for (int t = 0; t < NTILES; ++t) {
    const int n = (ng * NTILES + t) * 16 + (lane & 15);
    if (n >= Nact) continue;
    const float bv = bz[n];
#pragma unroll
    for (int mi = 0; mi < MTILES; ++mi) {
#pragma unroll
      for (int r = 0; r < 8; ++r) {
        const int m = (mtBase + mi) * 16 + sel * 8 + r;
        float v = acc[mi][t][r] + bv;
        if (RELU) v = fmaxf(v, 0.0f);
        if (OUT_F32) {
          ((float*)Cout)[(size_t)z * cStrideZ + (size_t)m * ldc + n] = v;
        } else {
          ((unsigned short*)Cout)[(size_t)z * cStrideZ + (size_t)m * ldc + n] =
              f2bf(v);
        }
      }
    }
  }
}

// ---------------------------------------------------------------------------
// Finalize: gate softmax, expert log_softmax, gated combine.
// d_out layout: output[B, N_CLS] then gate_outs[B, N_EXP] (f32).
// ---------------------------------------------------------------------------
__global__ void finalize_moe_k(const float* __restrict__ glog,
                               const float* __restrict__ elog,
                               float* __restrict__ out) {
  int b = blockIdx.x * blockDim.x + threadIdx.x;
  if (b >= BATCH) return;

  float g[N_EXP];
  float gmax = -3.0e38f;
#pragma unroll
  for (int e = 0; e < N_EXP; ++e) {
    g[e] = glog[(size_t)b * 16 + e];
    gmax = fmaxf(gmax, g[e]);
  }
  float gs = 0.0f;
#pragma unroll
  for (int e = 0; e < N_EXP; ++e) { g[e] = __expf(g[e] - gmax); gs += g[e]; }
  const float ginv = 1.0f / gs;

  float o[N_CLS];
#pragma unroll
  for (int c = 0; c < N_CLS; ++c) o[c] = 0.0f;

#pragma unroll
  for (int e = 0; e < N_EXP; ++e) {
    const float ge = g[e] * ginv;
    const float* lp = elog + ((size_t)e * BATCH + b) * 16;
    float l[N_CLS];
    float lmax = -3.0e38f;
#pragma unroll
    for (int c = 0; c < N_CLS; ++c) { l[c] = lp[c]; lmax = fmaxf(lmax, l[c]); }
    float s = 0.0f;
#pragma unroll
    for (int c = 0; c < N_CLS; ++c) s += __expf(l[c] - lmax);
    const float lse = lmax + __logf(s);
#pragma unroll
    for (int c = 0; c < N_CLS; ++c) o[c] += ge * (l[c] - lse);
  }

#pragma unroll
  for (int c = 0; c < N_CLS; ++c) out[(size_t)b * N_CLS + c] = o[c];
  float* gout = out + (size_t)BATCH * N_CLS;
#pragma unroll
  for (int e = 0; e < N_EXP; ++e) gout[(size_t)b * N_EXP + e] = g[e] * ginv;
}

// ---------------------------------------------------------------------------
extern "C" void kernel_launch(void* const* d_in, const int* in_sizes, int n_in,
                              void* d_out, int out_size, void* d_ws,
                              size_t ws_size, hipStream_t stream) {
  (void)in_sizes; (void)n_in; (void)out_size; (void)ws_size;

  const float* fs  = (const float*)d_in[0];
  const float* fp  = (const float*)d_in[1];
  const float* gW1 = (const float*)d_in[2];
  const float* gb1 = (const float*)d_in[3];
  const float* gW2 = (const float*)d_in[4];
  const float* gb2 = (const float*)d_in[5];
  const float* gW3 = (const float*)d_in[6];
  const float* gb3 = (const float*)d_in[7];
  const float* eW1 = (const float*)d_in[8];
  const float* eb1 = (const float*)d_in[9];
  const float* eW2 = (const float*)d_in[10];
  const float* eb2 = (const float*)d_in[11];
  const float* eW3 = (const float*)d_in[12];
  const float* eb3 = (const float*)d_in[13];
  float* out = (float*)d_out;

  // ---- workspace carve-out (256B aligned) ----
  char* ws = (char*)d_ws;
  size_t off = 0;
  auto alloc = [&](size_t bytes) -> char* {
    char* p = ws + off;
    off += (bytes + 255) & ~(size_t)255;
    return p;
  };
  unsigned short* Xb   = (unsigned short*)alloc((size_t)BATCH * D_IN * 2);
  unsigned short* h1g  = (unsigned short*)alloc((size_t)BATCH * H1 * 2);
  unsigned short* h1e  = (unsigned short*)alloc((size_t)N_EXP * BATCH * H1 * 2);
  unsigned short* h2g  = (unsigned short*)alloc((size_t)BATCH * H2 * 2);
  unsigned short* h2e  = (unsigned short*)alloc((size_t)N_EXP * BATCH * H2 * 2);
  float*          glog = (float*)alloc((size_t)BATCH * 16 * 4);
  float*          elog = (float*)alloc((size_t)N_EXP * BATCH * 16 * 4);
  unsigned short* pW1g = (unsigned short*)alloc((size_t)D_IN * H1 * 2);
  unsigned short* pW1e = (unsigned short*)alloc((size_t)N_EXP * D_IN * H1 * 2);
  unsigned short* pW2g = (unsigned short*)alloc((size_t)H1 * H2 * 2);
  unsigned short* pW2e = (unsigned short*)alloc((size_t)N_EXP * H1 * H2 * 2);
  unsigned short* pW3g = (unsigned short*)alloc((size_t)H2 * 16 * 2);
  unsigned short* pW3e = (unsigned short*)alloc((size_t)N_EXP * H2 * 16 * 2);

  // ---- pack features + weights (bf16, WMMA fragment order) ----
  {
    size_t tot = (size_t)BATCH * D_IN;
    pack_features_k<<<dim3((unsigned)((tot + 255) / 256)), dim3(256), 0, stream>>>(
        fs, fp, Xb);
  }
  auto packW = [&](const float* W, unsigned short* dst, int K, int N, int Npad,
                   int batch) {
    int tot = K * Npad;
    pack_weight_k<<<dim3((tot + 255) / 256, batch), dim3(256), 0, stream>>>(
        W, dst, K, N, Npad);
  };
  packW(gW1, pW1g, D_IN, H1, H1, 1);
  packW(eW1, pW1e, D_IN, H1, H1, N_EXP);
  packW(gW2, pW2g, H1, H2, H2, 1);
  packW(eW2, pW2e, H1, H2, H2, N_EXP);
  packW(gW3, pW3g, H2, N_EXP, 16, 1);
  packW(eW3, pW3e, H2, N_CLS, 16, N_EXP);

  // ---- GEMM stack ----
  const dim3 blk(256);
  const unsigned mBlocks2 = BATCH / (16 * 8 * 2);  // MTILES=2 -> 64
  const unsigned mBlocks1 = BATCH / (16 * 8 * 1);  // MTILES=1 -> 128

  // Layer 1: [B,2048] x [2048,512], ReLU, bf16 out
  gemm_bf16_wmma_k<2, 4, true, false>
      <<<dim3(mBlocks2, H1 / 64, 1), blk, 0, stream>>>(
          Xb, 0, D_IN, pW1g, 0, gb1, 0, h1g, 0, H1, H1, H1);
  gemm_bf16_wmma_k<2, 4, true, false>
      <<<dim3(mBlocks2, H1 / 64, N_EXP), blk, 0, stream>>>(
          Xb, 0, D_IN, pW1e, (size_t)D_IN * H1, eb1, H1, h1e,
          (size_t)BATCH * H1, H1, H1, H1);

  // Layer 2: [B,512] x [512,128], ReLU, bf16 out
  gemm_bf16_wmma_k<2, 4, true, false>
      <<<dim3(mBlocks2, H2 / 64, 1), blk, 0, stream>>>(
          h1g, 0, H1, pW2g, 0, gb2, 0, h2g, 0, H2, H2, H2);
  gemm_bf16_wmma_k<2, 4, true, false>
      <<<dim3(mBlocks2, H2 / 64, N_EXP), blk, 0, stream>>>(
          h1e, (size_t)BATCH * H1, H1, pW2e, (size_t)H1 * H2, eb2, H2, h2e,
          (size_t)BATCH * H2, H2, H2, H2);

  // Layer 3: [B,128] x [128,{8|10} pad 16], f32 logits out (ldc = 16)
  gemm_bf16_wmma_k<1, 1, false, true>
      <<<dim3(mBlocks1, 1, 1), blk, 0, stream>>>(
          h2g, 0, H2, pW3g, 0, gb3, 0, glog, 0, 16, 16, N_EXP);
  gemm_bf16_wmma_k<1, 1, false, true>
      <<<dim3(mBlocks1, 1, N_EXP), blk, 0, stream>>>(
          h2e, (size_t)BATCH * H2, H2, pW3e, (size_t)H2 * 16, eb3, N_CLS, elog,
          (size_t)BATCH * 16, 16, 16, N_CLS);

  // ---- softmax / log_softmax / gated combine ----
  finalize_moe_k<<<dim3((BATCH + 255) / 256), blk, 0, stream>>>(glog, elog, out);
}
